// MaskedGraphVAE_5583457485160
// MI455X (gfx1250) — matrix operand order repaired and checked by reading, not compiled, not run on hardware
//
#include <hip/hip_runtime.h>

typedef __attribute__((ext_vector_type(16))) _Float16 v16h;
typedef __attribute__((ext_vector_type(8)))  _Float16 v8h;
typedef __attribute__((ext_vector_type(8)))  float    v8f;

#define N_NODES  50000
#define N_EDGES  600000
#define N_GRAPHS 512
#define HID      128

union Frag16 { v16h v; v8h h8[2]; };
union Acc8   { v8f  v; float f[8]; };

// ---------------------------------------------------------------- utilities
__global__ void k_zero(float* __restrict__ p, int n) {
    int i = blockIdx.x * blockDim.x + threadIdx.x;
    if (i < n) p[i] = 0.0f;
}

// Pack conv weights (f32 [K=128][N=128]) into per-lane-contiguous f16 WMMA
// B-fragments.  Layout: [mat 0..5][tile t 0..7][lane n 0..31][p 0..63]
//   p = s*16 + q  (s = K-step of 32, q = 0..15)
//   K = s*32 + (n>=16 ? 16 : 0) + q,  col = t*16 + (n&15)
// so each lane's fragment for one K-step is one contiguous 32B v16h load.
__global__ void k_pack_weights(const float* __restrict__ w1,
                               const float* __restrict__ w2,
                               _Float16* __restrict__ bp) {
    int idx = blockIdx.x * blockDim.x + threadIdx.x;   // < 6*16384
    int p = idx & 63;
    int n = (idx >> 6) & 31;
    int t = (idx >> 11) & 7;
    int m = idx >> 14;                                  // 0..5
    int s = p >> 4, q = p & 15;
    int K   = s * 32 + ((n >= 16) ? 16 : 0) + q;
    int col = t * 16 + (n & 15);
    const float* W = (m < 3) ? (w1 + m * 16384) : (w2 + (m - 3) * 16384);
    bp[idx] = (_Float16)W[K * 128 + col];
}

// h = relu(x[:, :16] @ node_proj_w[:16, :] + b)  (mask columns are zero)
// 2 nodes / 256-thread block; x rows staged once in LDS.
__global__ void k_encode_nodes(const float* __restrict__ x,
                               const float* __restrict__ W,
                               const float* __restrict__ b,
                               float* __restrict__ h) {
    __shared__ float sX[32];
    int t = threadIdx.x;
    if (t < 32) sX[t] = x[blockIdx.x * 32 + t];
    __syncthreads();
    int nl = t >> 7, c = t & 127;
    float acc = b[c];
#pragma unroll
    for (int k = 0; k < 16; ++k) acc += sX[nl * 16 + k] * W[k * 128 + c];
    h[(size_t)(blockIdx.x * 2 + nl) * 128 + c] = fmaxf(acc, 0.0f);
}

// Per edge: e = relu(edge_attr @ We[:8,:] + be) (recomputed each layer,
// cheaper than 307MB of materialized edge embeddings), msg = relu(h[src]+e),
// agg[dst] += msg (float atomics -> global_atomic_add_f32).
// 2 edges / block; edge_attr + indices staged once in LDS.
__global__ void k_message(const float* __restrict__ ea,
                          const int* __restrict__ src,
                          const int* __restrict__ dst,
                          const float* __restrict__ We,
                          const float* __restrict__ be,
                          const float* __restrict__ h,
                          float* __restrict__ agg) {
    __shared__ float sEA[16];
    __shared__ int   sIdx[4];
    int t = threadIdx.x;
    if (t < 16) sEA[t] = ea[blockIdx.x * 16 + t];
    if (t < 2) {
        sIdx[t]     = src[blockIdx.x * 2 + t];
        sIdx[2 + t] = dst[blockIdx.x * 2 + t];
    }
    __syncthreads();
    int el = t >> 7, c = t & 127;
    float acc = be[c];
#pragma unroll
    for (int k = 0; k < 8; ++k) acc += sEA[el * 8 + k] * We[k * 128 + c];
    float ev = fmaxf(acc, 0.0f);
    int sN = sIdx[el], dN = sIdx[2 + el];
    float msg = fmaxf(h[(size_t)sN * 128 + c] + ev, 0.0f);
    atomicAdd(&agg[(size_t)dN * 128 + c], msg);
}

// Fused GINE MLP for a 64-row stripe:
//   hin = h + agg -> f16 LDS -> WMMA GEMM1 (+b1, relu) -> f16 LDS
//   -> WMMA GEMM2 (+b2, relu) -> hout (f32)
// 8 waves per block; wave w owns output columns [16w,16w+16) and iterates
// 4 M-tiles, reusing each B fragment across the 4 WMMAs (4x less L2 traffic).
__global__ void __launch_bounds__(256)
k_conv(const float* __restrict__ h, const float* __restrict__ agg,
       const _Float16* __restrict__ B1, const _Float16* __restrict__ B2,
       const float* __restrict__ b1, const float* __restrict__ b2,
       float* __restrict__ hout) {
    constexpr int ROWS = 64;
    constexpr int LDA  = 136;                 // pad 8 halves: no LDS bank conflicts
    __shared__ __align__(32) _Float16 sA[ROWS * LDA];
    __shared__ __align__(32) _Float16 sMid[ROWS * LDA];

    int t = threadIdx.x;
    int base = blockIdx.x * ROWS;
    int w = t >> 5, lane = t & 31;
    int m = lane & 15, hi = lane >> 4;
    int col = w * 16 + m;                     // C layout: N = lane%16

    // warm L2 with this lane's weight fragments (global_prefetch_b8)
    __builtin_prefetch(B1 + (w * 32 + lane) * 64, 0, 0);
    __builtin_prefetch(B2 + (w * 32 + lane) * 64, 0, 0);

    // stage hin = h + agg as f16 (64 x 128), float4 vector loads
#pragma unroll
    for (int i = 0; i < 8; ++i) {
        int v4 = t + i * 256;                 // 0..2047 float4s
        int r  = v4 >> 5;                     // 32 float4 per row
        int c4 = (v4 & 31) * 4;
        int row = base + r;
        if (row >= N_NODES) row = N_NODES - 1;
        float4 hv = *reinterpret_cast<const float4*>(&h[(size_t)row * 128 + c4]);
        float4 av = *reinterpret_cast<const float4*>(&agg[(size_t)row * 128 + c4]);
        _Float16* dp = &sA[r * LDA + c4];
        dp[0] = (_Float16)(hv.x + av.x);
        dp[1] = (_Float16)(hv.y + av.y);
        dp[2] = (_Float16)(hv.z + av.z);
        dp[3] = (_Float16)(hv.w + av.w);
    }
    __syncthreads();

    // ---- GEMM1: hin @ W1 ----
    Acc8 acc[4];
#pragma unroll
    for (int mt = 0; mt < 4; ++mt)
#pragma unroll
        for (int r = 0; r < 8; ++r) acc[mt].f[r] = 0.0f;
#pragma unroll
    for (int s = 0; s < 4; ++s) {
        v16h bf = *reinterpret_cast<const v16h*>(B1 + ((w * 32 + lane) * 64 + s * 16));
#pragma unroll
        for (int mt = 0; mt < 4; ++mt) {
            Frag16 a;   // A 16x32 f16: lanes<16 K{0..15}, lanes>=16 K{16..31}
            const _Float16* ar = &sA[(mt * 16 + m) * LDA + s * 32 + hi * 16];
            a.h8[0] = *reinterpret_cast<const v8h*>(ar);
            a.h8[1] = *reinterpret_cast<const v8h*>(ar + 8);
            acc[mt].v = __builtin_amdgcn_wmma_f32_16x16x32_f16(
                false, a.v, false, bf, (short)0, acc[mt].v, false, false);
        }
    }
    float bias1 = b1[col];
#pragma unroll
    for (int mt = 0; mt < 4; ++mt)
#pragma unroll
        for (int r = 0; r < 8; ++r) {
            float f = fmaxf(acc[mt].f[r] + bias1, 0.0f);   // C: M = r + 8*hi
            sMid[(mt * 16 + r + hi * 8) * LDA + col] = (_Float16)f;
        }
    __syncthreads();

    // ---- GEMM2: hmid @ W2 ----
#pragma unroll
    for (int mt = 0; mt < 4; ++mt)
#pragma unroll
        for (int r = 0; r < 8; ++r) acc[mt].f[r] = 0.0f;
#pragma unroll
    for (int s = 0; s < 4; ++s) {
        v16h bf = *reinterpret_cast<const v16h*>(B2 + ((w * 32 + lane) * 64 + s * 16));
#pragma unroll
        for (int mt = 0; mt < 4; ++mt) {
            Frag16 a;
            const _Float16* ar = &sMid[(mt * 16 + m) * LDA + s * 32 + hi * 16];
            a.h8[0] = *reinterpret_cast<const v8h*>(ar);
            a.h8[1] = *reinterpret_cast<const v8h*>(ar + 8);
            acc[mt].v = __builtin_amdgcn_wmma_f32_16x16x32_f16(
                false, a.v, false, bf, (short)0, acc[mt].v, false, false);
        }
    }
    float bias2 = b2[col];
#pragma unroll
    for (int mt = 0; mt < 4; ++mt)
#pragma unroll
        for (int r = 0; r < 8; ++r) {
            int row = base + mt * 16 + r + hi * 8;
            if (row < N_NODES) {
                float f = fmaxf(acc[mt].f[r] + bias2, 0.0f);
                hout[(size_t)row * 128 + col] = f;
            }
        }
}

// Scatter pooling: mean via atomicAdd, max via uint atomicMax (h >= 0 post-relu
// so IEEE order == unsigned order), counts via atomicAdd on channel 0.
__global__ void k_pool(const float* __restrict__ h, const int* __restrict__ batch,
                       float* __restrict__ meanAcc, unsigned* __restrict__ maxBits,
                       float* __restrict__ cnt) {
    int idx = blockIdx.x * blockDim.x + threadIdx.x;    // N_NODES*128
    int n = idx >> 7, c = idx & 127;
    int g = batch[n];
    float v = h[idx];
    atomicAdd(&meanAcc[g * 128 + c], v);
    atomicMax(&maxBits[g * 128 + c], __float_as_uint(v));
    if (c == 0) atomicAdd(&cnt[g], 1.0f);
}

// Per-graph head: pooled[256] -> mu[8] -> relu(mu@aw1+ab1)@aw2+ab2 -> out[g][9]
__global__ void __launch_bounds__(128)
k_head(const float* __restrict__ meanAcc, const unsigned* __restrict__ maxBits,
       const float* __restrict__ cnt,
       const float* __restrict__ muw, const float* __restrict__ mub,
       const float* __restrict__ aw1, const float* __restrict__ ab1,
       const float* __restrict__ aw2, const float* __restrict__ ab2,
       float* __restrict__ out) {
    int g = blockIdx.x, t = threadIdx.x;                // 128 threads
    __shared__ float pooled[256];
    __shared__ float smu[8];
    __shared__ float red[128];
    float inv = 1.0f / fmaxf(cnt[g], 1.0f);
    pooled[t]       = meanAcc[g * 128 + t] * inv;
    pooled[128 + t] = __uint_as_float(maxBits[g * 128 + t]);
    __syncthreads();
    if (t < 8) {
        float a = mub[t];
        for (int i = 0; i < 256; ++i) a += pooled[i] * muw[i * 8 + t];
        smu[t] = a;
    }
    __syncthreads();
    float hv = ab1[t];
#pragma unroll
    for (int j = 0; j < 8; ++j) hv += smu[j] * aw1[j * 128 + t];
    hv = fmaxf(hv, 0.0f);
    red[t] = hv * aw2[t];
    __syncthreads();
    for (int s = 64; s > 0; s >>= 1) {
        if (t < s) red[t] += red[t + s];
        __syncthreads();
    }
    if (t < 8)  out[g * 9 + t] = smu[t];
    if (t == 0) out[g * 9 + 8] = red[0] + ab2[0];       // AFF_STD=1, AFF_MEAN=0
}

extern "C" void kernel_launch(void* const* d_in, const int* in_sizes, int n_in,
                              void* d_out, int out_size, void* d_ws, size_t ws_size,
                              hipStream_t stream) {
    (void)in_sizes; (void)n_in; (void)out_size; (void)ws_size;
    const float* x     = (const float*)d_in[0];
    const float* ea    = (const float*)d_in[1];
    const int*   ei    = (const int*)d_in[2];
    const int*   batch = (const int*)d_in[3];
    const float* npw   = (const float*)d_in[4];
    const float* npb   = (const float*)d_in[5];
    const float* epw   = (const float*)d_in[6];
    const float* epb   = (const float*)d_in[7];
    const float* cw1   = (const float*)d_in[8];
    const float* cb1   = (const float*)d_in[9];
    const float* cw2   = (const float*)d_in[10];
    const float* cb2   = (const float*)d_in[11];
    const float* muw   = (const float*)d_in[12];
    const float* mub   = (const float*)d_in[13];
    const float* aw1   = (const float*)d_in[14];
    const float* ab1   = (const float*)d_in[15];
    const float* aw2   = (const float*)d_in[16];
    const float* ab2   = (const float*)d_in[17];
    float* out = (float*)d_out;

    // workspace layout (floats): hA | hB | agg | meanAcc | maxBits | cnt | bpack(f16)
    float* ws      = (float*)d_ws;
    float* hA      = ws;
    float* hB      = hA + (size_t)N_NODES * HID;
    float* agg     = hB + (size_t)N_NODES * HID;
    float* meanAcc = agg + (size_t)N_NODES * HID;
    unsigned* maxBits = (unsigned*)(meanAcc + N_GRAPHS * HID);
    float* cnt     = (float*)(maxBits + N_GRAPHS * HID);
    _Float16* bpack = (_Float16*)(cnt + 512);           // 32B-aligned by construction

    const int* src = ei;
    const int* dst = ei + N_EDGES;

    k_pack_weights<<<384, 256, 0, stream>>>(cw1, cw2, bpack);               // 6*16384 elems
    k_zero<<<(N_GRAPHS * HID * 2 + 512 + 255) / 256, 256, 0, stream>>>(
        meanAcc, N_GRAPHS * HID * 2 + 512);                                  // pool accumulators
    k_encode_nodes<<<N_NODES / 2, 256, 0, stream>>>(x, npw, npb, hA);

    float* hcur = hA, *hnext = hB;
    for (int l = 0; l < 3; ++l) {
        k_zero<<<N_NODES * HID / 256, 256, 0, stream>>>(agg, N_NODES * HID);
        k_message<<<N_EDGES / 2, 256, 0, stream>>>(ea, src, dst, epw, epb, hcur, agg);
        k_conv<<<(N_NODES + 63) / 64, 256, 0, stream>>>(hcur, agg,
                                                        bpack + l * 16384,
                                                        bpack + (3 + l) * 16384,
                                                        cb1 + l * HID, cb2 + l * HID,
                                                        hnext);
        float* tmp = hcur; hcur = hnext; hnext = tmp;
    }
    k_pool<<<N_NODES * HID / 256, 256, 0, stream>>>(hcur, batch, meanAcc, maxBits, cnt);
    k_head<<<N_GRAPHS, 128, 0, stream>>>(meanAcc, maxBits, cnt,
                                         muw, mub, aw1, ab1, aw2, ab2, out);
}
